// GraphSAGEEncoder_712964571452
// MI455X (gfx1250) — matrix operand order repaired and checked by reading, not compiled, not run on hardware
//
#include <hip/hip_runtime.h>
#include <hip/hip_bf16.h>

typedef __attribute__((ext_vector_type(16))) _Float16 v16h;
typedef __attribute__((ext_vector_type(8)))  float    v8f;

#define NNODES 100000
#define DFEAT  64

// ---------------------------------------------------------------------------
// utility: zero a float buffer (float4 granularity)
// ---------------------------------------------------------------------------
__global__ void k_zero4(float* __restrict__ p, int n4) {
  int i = blockIdx.x * blockDim.x + threadIdx.x;
  if (i < n4) ((float4*)p)[i] = make_float4(0.f, 0.f, 0.f, 0.f);
}

// ---------------------------------------------------------------------------
// degree count: one thread per edge, atomicAdd 1.0 to deg[dst]
// ---------------------------------------------------------------------------
__global__ void k_deg(const long long* __restrict__ dst, int E,
                      float* __restrict__ deg) {
  int e = blockIdx.x * blockDim.x + threadIdx.x;
  if (e < E) atomicAdd(&deg[(int)dst[e]], 1.0f);
}

__global__ void k_invdeg(float* __restrict__ deg, int n) {
  int i = blockIdx.x * blockDim.x + threadIdx.x;
  if (i < n) deg[i] = 1.0f / fmaxf(deg[i], 1.0f);
}

// ---------------------------------------------------------------------------
// edge scatter: 16 threads per edge, each moves a float4 chunk of the 64-dim
// message and atomically accumulates into agg[dst]. agg (25.6MB) is L2-resident.
// ---------------------------------------------------------------------------
__global__ void k_scatter(const float* __restrict__ x,
                          const long long* __restrict__ src,
                          const long long* __restrict__ dst, int E,
                          float* __restrict__ agg) {
  long long gid = (long long)blockIdx.x * blockDim.x + threadIdx.x;
  int e = (int)(gid >> 4);
  if (e >= E) return;
  int c = ((int)gid & 15) << 2;            // feature chunk: 0,4,...,60
  int s = (int)src[e];
  int d = (int)dst[e];
  float4 v = *(const float4*)(x + (size_t)s * DFEAT + c);
  float* a = agg + (size_t)d * DFEAT + c;
  atomicAdd(a + 0, v.x);
  atomicAdd(a + 1, v.y);
  atomicAdd(a + 2, v.z);
  atomicAdd(a + 3, v.w);
}

// ---------------------------------------------------------------------------
// WMMA helpers
// ---------------------------------------------------------------------------
__device__ __forceinline__ v8f wmma16(v16h a, v16h b, v8f c) {
  // (neg_a, A, neg_b, B, c_mod, C, reuse_a, reuse_b)
  return __builtin_amdgcn_wmma_f32_16x16x32_f16(false, a, false, b,
                                                (short)0, c, false, false);
}

// A fragment (16x32 f16): lane l -> row (l&15); K runs: h*16+0..15
__device__ __forceinline__ v16h load_a_frag(const float* __restrict__ base,
                                            int m, int h, int kb, float scale) {
  const float* p = base + (size_t)m * DFEAT + kb + h * 16;
  float4 f0 = *(const float4*)(p + 0);
  float4 f1 = *(const float4*)(p + 4);
  float4 f2 = *(const float4*)(p + 8);
  float4 f3 = *(const float4*)(p + 12);
  v16h a;
  a[0]  = (_Float16)(f0.x * scale); a[1]  = (_Float16)(f0.y * scale);
  a[2]  = (_Float16)(f0.z * scale); a[3]  = (_Float16)(f0.w * scale);
  a[4]  = (_Float16)(f1.x * scale); a[5]  = (_Float16)(f1.y * scale);
  a[6]  = (_Float16)(f1.z * scale); a[7]  = (_Float16)(f1.w * scale);
  a[8]  = (_Float16)(f2.x * scale); a[9]  = (_Float16)(f2.y * scale);
  a[10] = (_Float16)(f2.z * scale); a[11] = (_Float16)(f2.w * scale);
  a[12] = (_Float16)(f3.x * scale); a[13] = (_Float16)(f3.y * scale);
  a[14] = (_Float16)(f3.z * scale); a[15] = (_Float16)(f3.w * scale);
  return a;
}

// ---------------------------------------------------------------------------
// fused SAGE layer GEMM:
//   out[m][:] = act( (agg[m]*invdeg[m]) @ Wl + x[m] @ Wr + bl ) (+ x[m] if res)
// 8 waves / block, each wave owns a 16-row strip and all 64 output columns
// (4 x v8f accumulators). Weights transposed to f16 in LDS so each B fragment
// is one contiguous 32B read (B order used here, same k as A: lane l -> col l&15,
// K = (l>>4)*16 + 0..15).
// ---------------------------------------------------------------------------
__global__ __launch_bounds__(256) void k_sage_gemm(
    const float* __restrict__ xin, const float* __restrict__ agg,
    const float* __restrict__ invdeg,
    const float* __restrict__ Wl, const float* __restrict__ bl,
    const float* __restrict__ Wr,
    float* __restrict__ out, int relu, int residual) {
  __shared__ __align__(32) _Float16 sWl[DFEAT * DFEAT];  // transposed [n][k]
  __shared__ __align__(32) _Float16 sWr[DFEAT * DFEAT];
  __shared__ float sb[DFEAT];

  int t = threadIdx.x;
#pragma unroll
  for (int u = 0; u < 16; ++u) {
    int idx = t * 16 + u;                 // 256*16 == 64*64
    int k = idx >> 6, n = idx & 63;
    sWl[n * DFEAT + k] = (_Float16)Wl[idx];
    sWr[n * DFEAT + k] = (_Float16)Wr[idx];
  }
  if (t < DFEAT) sb[t] = bl[t];
  __syncthreads();

  int wave = t >> 5, lane = t & 31;
  int strip = blockIdx.x * 8 + wave;
  if (strip * 16 >= NNODES) return;       // wave-uniform: EXEC stays all-ones
  int m0 = strip * 16;
  int lr = lane & 15, h = lane >> 4;

  float sc = invdeg[m0 + lr];             // scale for this lane's A rows

  v8f acc[4] = {v8f{}, v8f{}, v8f{}, v8f{}};

#pragma unroll
  for (int kt = 0; kt < 2; ++kt) {
    int kb = kt * 32;
    v16h aAgg = load_a_frag(agg, m0 + lr, h, kb, sc);
    v16h aX   = load_a_frag(xin, m0 + lr, h, kb, 1.0f);
#pragma unroll
    for (int nt = 0; nt < 4; ++nt) {
      const _Float16* pl = sWl + (nt * 16 + lr) * DFEAT + kb + h * 16;
      const _Float16* pr = sWr + (nt * 16 + lr) * DFEAT + kb + h * 16;
      v16h bL = *(const v16h*)pl;
      v16h bR = *(const v16h*)pr;
      acc[nt] = wmma16(aAgg, bL, acc[nt]);
      acc[nt] = wmma16(aX,   bR, acc[nt]);
    }
  }

  // epilogue: bias + relu + residual; C layout: vgpr j -> row h*8+j, col lr
#pragma unroll
  for (int nt = 0; nt < 4; ++nt) {
    int nc = nt * 16 + lr;
    float bb = sb[nc];
#pragma unroll
    for (int j = 0; j < 8; ++j) {
      int m = m0 + h * 8 + j;
      float v = acc[nt][j] + bb;
      if (relu) v = fmaxf(v, 0.0f);
      if (residual) v += xin[(size_t)m * DFEAT + nc];
      out[(size_t)m * DFEAT + nc] = v;
    }
  }
}

// ---------------------------------------------------------------------------
// row-wise L2 normalize: one wave32 per row, 2 elems/lane, shfl_xor reduce
// ---------------------------------------------------------------------------
__global__ __launch_bounds__(256) void k_norm(float* __restrict__ out, int nrows) {
  int wave = threadIdx.x >> 5, lane = threadIdx.x & 31;
  int row = blockIdx.x * 8 + wave;
  if (row >= nrows) return;
  float* p = out + (size_t)row * DFEAT;
  float a = p[lane];
  float b = p[lane + 32];
  float s = a * a + b * b;
#pragma unroll
  for (int m = 16; m >= 1; m >>= 1) s += __shfl_xor(s, m, 32);
  float inv = 1.0f / fmaxf(sqrtf(s), 1e-12f);
  p[lane] = a * inv;
  p[lane + 32] = b * inv;
}

// ---------------------------------------------------------------------------
// host launcher
// ---------------------------------------------------------------------------
extern "C" void kernel_launch(void* const* d_in, const int* in_sizes, int n_in,
                              void* d_out, int out_size, void* d_ws, size_t ws_size,
                              hipStream_t stream) {
  const float* x = (const float*)d_in[0];
  const long long* ei = (const long long*)d_in[1];
  const int E = in_sizes[1] / 2;
  const long long* srcv = ei;
  const long long* dstv = ei + E;

  const float* Wl[4] = {(const float*)d_in[2], (const float*)d_in[5],
                        (const float*)d_in[8], (const float*)d_in[11]};
  const float* bl[4] = {(const float*)d_in[3], (const float*)d_in[6],
                        (const float*)d_in[9], (const float*)d_in[12]};
  const float* Wr[4] = {(const float*)d_in[4], (const float*)d_in[7],
                        (const float*)d_in[10], (const float*)d_in[13]};

  const size_t FSZ = (size_t)NNODES * DFEAT * sizeof(float);  // 25.6 MB
  char* ws = (char*)d_ws;
  float* deg = (float*)ws;                          // N floats (-> inv_deg)
  float* agg = (float*)(ws + (1 << 20));
  float* hA  = (float*)(ws + (1 << 20) + FSZ);
  float* hB  = (float*)(ws + (1 << 20) + 2 * FSZ);
  float* outp = (float*)d_out;

  const int THR = 256;
  const int n4_deg = NNODES / 4;                    // 25000
  const int n4_agg = NNODES * DFEAT / 4;            // 1,600,000
  const long long scat_thr = (long long)E * 16;
  const int scat_blocks = (int)((scat_thr + THR - 1) / THR);
  const int gemm_blocks = ((NNODES + 15) / 16 + 7) / 8;   // 782
  const int norm_blocks = (NNODES + 7) / 8;

  // degrees (same for all layers)
  k_zero4<<<(n4_deg + THR - 1) / THR, THR, 0, stream>>>(deg, n4_deg);
  k_deg<<<(E + THR - 1) / THR, THR, 0, stream>>>(dstv, E, deg);
  k_invdeg<<<(NNODES + THR - 1) / THR, THR, 0, stream>>>(deg, NNODES);

  struct Layer { const float* in; float* out; int relu; int res; } L[4] = {
      {x,  hA,   1, 0},
      {hA, hB,   1, 1},
      {hB, hA,   1, 1},
      {hA, outp, 0, 0},
  };

  for (int i = 0; i < 4; ++i) {
    k_zero4<<<(n4_agg + THR - 1) / THR, THR, 0, stream>>>(agg, n4_agg);
    k_scatter<<<scat_blocks, THR, 0, stream>>>(L[i].in, srcv, dstv, E, agg);
    k_sage_gemm<<<gemm_blocks, THR, 0, stream>>>(L[i].in, agg, deg,
                                                 Wl[i], bl[i], Wr[i],
                                                 L[i].out, L[i].relu, L[i].res);
  }

  k_norm<<<norm_blocks, THR, 0, stream>>>(outp, NNODES);
}
